// BiRNN_CRF_12120397709946
// MI455X (gfx1250) — matrix operand order repaired and checked by reading, not compiled, not run on hardware
//
#include <hip/hip_runtime.h>
#include <hip/hip_bf16.h>

// ---------------------------------------------------------------------------
// BiLSTM-CRF for MI455X (gfx1250, wave32, WMMA bf16 16x16x32)
// B=32, S=512, E=300, H=512, T=9
// ---------------------------------------------------------------------------

#define BB    32
#define SS    512
#define EEIN  300
#define EPAD  320      // E padded to multiple of 32 for WMMA K
#define HH    512
#define G4H   2048     // 4*H gate rows
#define TT    9

typedef __attribute__((ext_vector_type(16))) __bf16 v16bf;
typedef __attribute__((ext_vector_type(8)))  float  v8f;

union FragU { v16bf v; uint4 q[2]; };

// ---- WMMA fragment loads (wave32, 16x16x32 bf16) --------------------------
// A (16M x 32K), row-major memory [row][k], ld multiple of 8 elements:
//   lane L: row = L%16, half = L/16
//   VGPR0-3 hold K = half*8 .. half*8+7      (16 contiguous bytes)
//   VGPR4-7 hold K = 16+half*8 .. 16+half*8+7 (16 contiguous bytes)
__device__ __forceinline__ v16bf load_fragA(const __hip_bfloat16* base,
                                            int ld, int k0, int lane) {
  int row  = lane & 15;
  int half = lane >> 4;
  const __hip_bfloat16* p = base + (size_t)row * ld + k0 + half * 8;
  FragU f;
  f.q[0] = *(const uint4*)(p);
  f.q[1] = *(const uint4*)(p + 16);
  return f.v;
}

// B (32K x 16N) from row-major [n][k] memory (i.e. weight[gate_row][k]):
//   lane L: n = L%16, half = L/16
//   VGPR0-3 hold K = half*8 .. half*8+7, VGPR4-7 hold K = 16+half*8 .. 16+half*8+7
__device__ __forceinline__ v16bf load_fragB(const __hip_bfloat16* base,
                                            int ld, int n0, int k0, int lane) {
  int n    = n0 + (lane & 15);
  int half = lane >> 4;
  const __hip_bfloat16* p = base + (size_t)n * ld + k0 + half * 8;
  FragU f;
  f.q[0] = *(const uint4*)(p);
  f.q[1] = *(const uint4*)(p + 16);
  return f.v;
}

__device__ __forceinline__ v8f wmma_bf16(v16bf a, v16bf b, v8f c) {
  return __builtin_amdgcn_wmma_f32_16x16x32_bf16(
      /*neg_a=*/false, a, /*neg_b=*/false, b,
      /*c_mod=*/(short)0, c, /*reuse_a=*/false, /*reuse_b=*/false);
}

__device__ __forceinline__ float sigm_f(float x) {
  return 1.0f / (1.0f + __expf(-x));
}
__device__ __forceinline__ float tanh_f(float x) {
  return 1.0f - 2.0f / (__expf(2.0f * x) + 1.0f);
}

// ---- device-wide generation barrier (persistent-kernel scan sync) ---------
__device__ __forceinline__ void grid_barrier(int* bar, int nwg) {
  __threadfence();
  __syncthreads();
  if (threadIdx.x == 0) {
    int gen = __hip_atomic_load(&bar[1], __ATOMIC_RELAXED, __HIP_MEMORY_SCOPE_AGENT);
    int ticket = __hip_atomic_fetch_add(&bar[0], 1, __ATOMIC_ACQ_REL, __HIP_MEMORY_SCOPE_AGENT);
    if (ticket == nwg - 1) {
      __hip_atomic_store(&bar[0], 0, __ATOMIC_RELAXED, __HIP_MEMORY_SCOPE_AGENT);
      __hip_atomic_store(&bar[1], gen + 1, __ATOMIC_RELEASE, __HIP_MEMORY_SCOPE_AGENT);
    } else {
      while (__hip_atomic_load(&bar[1], __ATOMIC_ACQUIRE, __HIP_MEMORY_SCOPE_AGENT) == gen) {
        __builtin_amdgcn_s_sleep(2);
      }
    }
  }
  __syncthreads();
}

// ---------------------------------------------------------------------------
// Packing kernels: fp32 -> bf16, K padded with zeros, layouts for WMMA loads
// ---------------------------------------------------------------------------
__global__ void pack_x0_kernel(const float* __restrict__ x,        // [B][S][E]
                               __hip_bfloat16* __restrict__ dst) { // [S][B][EPAD]
  int idx = blockIdx.x * blockDim.x + threadIdx.x;
  const int total = SS * BB * EPAD;
  if (idx >= total) return;
  int s = idx / (BB * EPAD);
  int r = idx % (BB * EPAD);
  int b = r / EPAD;
  int k = r % EPAD;
  float v = (k < EEIN) ? x[((size_t)b * SS + s) * EEIN + k] : 0.0f;
  dst[idx] = __float2bfloat16(v);
}

__global__ void pack_rows_kernel(const float* __restrict__ src,    // [rows][kin]
                                 __hip_bfloat16* __restrict__ dst, // [rows][kpad]
                                 int rows, int kin, int kpad) {
  int idx = blockIdx.x * blockDim.x + threadIdx.x;
  int total = rows * kpad;
  if (idx >= total) return;
  int r = idx / kpad;
  int k = idx % kpad;
  float v = (k < kin) ? src[(size_t)r * kin + k] : 0.0f;
  dst[idx] = __float2bfloat16(v);
}

__global__ void bias_sum_kernel(const float* __restrict__ bi,
                                const float* __restrict__ bh,
                                float* __restrict__ out, int n) {
  int i = blockIdx.x * blockDim.x + threadIdx.x;
  if (i < n) out[i] = bi[i] + bh[i];
}

__global__ void init_bar_kernel(int* bar) {
  if (threadIdx.x < 8) bar[threadIdx.x] = 0;
}

// ---------------------------------------------------------------------------
// Persistent BiLSTM layer: both directions in one grid.
// 128 waves total = dir(2) x mtile(2 batch tiles of 16) x jtile(32 hidden tiles)
// Each wave computes all 4 gates for its 16x16 (batch, hidden) tile so the
// LSTM cell update is done fully in-register; one grid barrier per timestep.
//   gates = x_t @ Wih^T + h_{t-1} @ Whh^T + (b_ih + b_hh)      (fused)
// ---------------------------------------------------------------------------
__global__ __launch_bounds__(256)
void lstm_layer_kernel(const __hip_bfloat16* __restrict__ X,   // [S][B][Kx]
                       int Kx,
                       const __hip_bfloat16* __restrict__ WIH, // [2][2048][Kx]
                       const __hip_bfloat16* __restrict__ WHH, // [2][2048][512]
                       const float* __restrict__ BIAS,         // [2][2048]
                       __hip_bfloat16* __restrict__ Hout,      // [S][B][1024]
                       float* __restrict__ Cst,                // [2][32][512]
                       int* bar) {
  const int lane  = threadIdx.x & 31;
  const int gw    = blockIdx.x * (blockDim.x >> 5) + (threadIdx.x >> 5);
  const int dir   = gw >> 6;        // 0 = forward, 1 = backward
  const int rr    = gw & 63;
  const int mtile = rr & 1;         // batch tile (rows 0-15 / 16-31)
  const int jtile = rr >> 1;        // hidden tile 0..31
  const int half  = lane >> 4;
  const int nlane = lane & 15;
  const int jcol  = jtile * 16 + nlane;

  const __hip_bfloat16* wih = WIH + (size_t)dir * G4H * Kx;
  const __hip_bfloat16* whh = WHH + (size_t)dir * G4H * HH;
  const float b_i = BIAS[dir * G4H + 0 * HH + jcol];
  const float b_f = BIAS[dir * G4H + 1 * HH + jcol];
  const float b_g = BIAS[dir * G4H + 2 * HH + jcol];
  const float b_o = BIAS[dir * G4H + 3 * HH + jcol];
  float* cptr = Cst + (size_t)dir * BB * HH;

  const v8f vzero = {0.f, 0.f, 0.f, 0.f, 0.f, 0.f, 0.f, 0.f};

  for (int t = 0; t < SS; ++t) {
    const int s_io = dir ? (SS - 1 - t) : t;   // input & output timestep
    v8f accI = vzero, accF = vzero, accG = vzero, accO = vzero;

    // ---- x_t @ Wih^T ------------------------------------------------------
    const __hip_bfloat16* xbase = X + ((size_t)s_io * BB + mtile * 16) * Kx;
    for (int kc = 0; kc < Kx; kc += 32) {
      v16bf a = load_fragA(xbase, Kx, kc, lane);
      v16bf bI = load_fragB(wih, Kx, 0 * HH + jtile * 16, kc, lane);
      accI = wmma_bf16(a, bI, accI);
      v16bf bF = load_fragB(wih, Kx, 1 * HH + jtile * 16, kc, lane);
      accF = wmma_bf16(a, bF, accF);
      v16bf bG = load_fragB(wih, Kx, 2 * HH + jtile * 16, kc, lane);
      accG = wmma_bf16(a, bG, accG);
      v16bf bO = load_fragB(wih, Kx, 3 * HH + jtile * 16, kc, lane);
      accO = wmma_bf16(a, bO, accO);
    }

    // ---- h_{t-1} @ Whh^T (skipped at t==0: h0 = 0) ------------------------
    if (t > 0) {
      const int s_prev = dir ? (s_io + 1) : (s_io - 1);
      const __hip_bfloat16* hbase =
          Hout + ((size_t)s_prev * BB + mtile * 16) * (2 * HH) + dir * HH;
      for (int kc = 0; kc < HH; kc += 32) {
        v16bf a = load_fragA(hbase, 2 * HH, kc, lane);
        v16bf bI = load_fragB(whh, HH, 0 * HH + jtile * 16, kc, lane);
        accI = wmma_bf16(a, bI, accI);
        v16bf bF = load_fragB(whh, HH, 1 * HH + jtile * 16, kc, lane);
        accF = wmma_bf16(a, bF, accF);
        v16bf bG = load_fragB(whh, HH, 2 * HH + jtile * 16, kc, lane);
        accG = wmma_bf16(a, bG, accG);
        v16bf bO = load_fragB(whh, HH, 3 * HH + jtile * 16, kc, lane);
        accO = wmma_bf16(a, bO, accO);
      }
    }

    // ---- in-register LSTM cell update ------------------------------------
    // C/D layout: VGPR r, lane L -> row M = r + 8*(L/16), col N = L%16
    for (int r = 0; r < 8; ++r) {
      const int brow = mtile * 16 + half * 8 + r;
      float gi = accI[r] + b_i;
      float gf = accF[r] + b_f;
      float gg = accG[r] + b_g;
      float go = accO[r] + b_o;
      float cprev = (t == 0) ? 0.0f : cptr[(size_t)brow * HH + jcol];
      float cn = sigm_f(gf) * cprev + sigm_f(gi) * tanh_f(gg);
      float hn = sigm_f(go) * tanh_f(cn);
      cptr[(size_t)brow * HH + jcol] = cn;
      Hout[((size_t)s_io * BB + brow) * (2 * HH) + dir * HH + jcol] =
          __float2bfloat16(hn);
    }

    grid_barrier(bar, gridDim.x);
  }
}

// ---------------------------------------------------------------------------
// Projection: emissions[b][s][t] = H1[s][b][:] . proj_w[t][:] + proj_b[t]
// ---------------------------------------------------------------------------
__global__ void proj_kernel(const __hip_bfloat16* __restrict__ H1, // [S][B][1024]
                            const float* __restrict__ pw,          // [9][1024]
                            const float* __restrict__ pb,          // [9]
                            float* __restrict__ emit) {            // [B][S][9]
  int idx = blockIdx.x * blockDim.x + threadIdx.x;
  const int total = BB * SS * TT;
  if (idx >= total) return;
  int b  = idx / (SS * TT);
  int r  = idx % (SS * TT);
  int s  = r / TT;
  int tt = r % TT;
  const __hip_bfloat16* h = H1 + ((size_t)s * BB + b) * (2 * HH);
  const float* w = pw + (size_t)tt * (2 * HH);
  float acc = 0.0f;
#pragma unroll 4
  for (int k = 0; k < 2 * HH; ++k) acc += __bfloat162float(h[k]) * w[k];
  emit[((size_t)b * SS + s) * TT + tt] = acc + pb[tt];
}

// ---------------------------------------------------------------------------
// CRF NLL: numerator (gold path score) + denominator (forward algorithm)
// T=9 is tiny -> one thread per batch element, block-level mean reduce.
// ---------------------------------------------------------------------------
__global__ void crf_kernel(const float* __restrict__ emit,  // [B][S][9]
                           const int* __restrict__ tags,    // [B][S]
                           const int* __restrict__ mask,    // [B][S]
                           const float* __restrict__ trans, // [9][9]
                           const float* __restrict__ start, // [9]
                           const float* __restrict__ endt,  // [9]
                           float* __restrict__ out) {
  __shared__ float res[BB];
  int b = threadIdx.x;
  if (b < BB) {
    const float* eb = emit + (size_t)b * SS * TT;
    const int* tb = tags + (size_t)b * SS;
    const int* mb = mask + (size_t)b * SS;

    // numerator
    int t0 = tb[0];
    float num = start[t0] + eb[t0];
    int cnt = (mb[0] != 0) ? 1 : 0;
    for (int s = 1; s < SS; ++s) {
      float m = (float)mb[s];
      num += (eb[(size_t)s * TT + tb[s]] + trans[tb[s - 1] * TT + tb[s]]) * m;
      cnt += (mb[s] != 0) ? 1 : 0;
    }
    int last_idx = cnt - 1;
    num += endt[tb[last_idx]];

    // denominator: forward algorithm
    float alpha[TT];
    for (int j = 0; j < TT; ++j) alpha[j] = start[j] + eb[j];
    for (int s = 1; s < SS; ++s) {
      if (mb[s]) {
        float na[TT];
        for (int j = 0; j < TT; ++j) {
          float mx = -1e30f;
          for (int i = 0; i < TT; ++i)
            mx = fmaxf(mx, alpha[i] + trans[i * TT + j]);
          float sm = 0.0f;
          for (int i = 0; i < TT; ++i)
            sm += __expf(alpha[i] + trans[i * TT + j] - mx);
          na[j] = mx + __logf(sm) + eb[(size_t)s * TT + j];
        }
        for (int j = 0; j < TT; ++j) alpha[j] = na[j];
      }
    }
    float mx = -1e30f;
    for (int j = 0; j < TT; ++j) mx = fmaxf(mx, alpha[j] + endt[j]);
    float sm = 0.0f;
    for (int j = 0; j < TT; ++j) sm += __expf(alpha[j] + endt[j] - mx);
    float denom = mx + __logf(sm);
    res[b] = num - denom;
  }
  __syncthreads();
  if (threadIdx.x == 0) {
    float s = 0.0f;
    for (int i = 0; i < BB; ++i) s += res[i];
    out[0] = -s / (float)BB;
  }
}

// ---------------------------------------------------------------------------
extern "C" void kernel_launch(void* const* d_in, const int* in_sizes, int n_in,
                              void* d_out, int out_size, void* d_ws, size_t ws_size,
                              hipStream_t stream) {
  const float* word_emb = (const float*)d_in[0];   // [32][512][300]
  const int*   tags     = (const int*)d_in[1];     // [32][512]
  const int*   amask    = (const int*)d_in[2];     // [32][512]
  const float* wih0     = (const float*)d_in[3];   // [2][2048][300]
  const float* whh0     = (const float*)d_in[4];   // [2][2048][512]
  const float* bih0     = (const float*)d_in[5];   // [2][2048]
  const float* bhh0     = (const float*)d_in[6];   // [2][2048]
  const float* wih1     = (const float*)d_in[7];   // [2][2048][1024]
  const float* whh1     = (const float*)d_in[8];   // [2][2048][512]
  const float* bih1     = (const float*)d_in[9];   // [2][2048]
  const float* bhh1     = (const float*)d_in[10];  // [2][2048]
  const float* pw       = (const float*)d_in[11];  // [9][1024]
  const float* pb       = (const float*)d_in[12];  // [9]
  const float* trans    = (const float*)d_in[13];  // [9][9]
  const float* startt   = (const float*)d_in[14];  // [9]
  const float* endt     = (const float*)d_in[15];  // [9]
  float* out = (float*)d_out;

  // ---- workspace carve ----
  char* ws = (char*)d_ws;
  size_t off = 0;
  auto carve = [&](size_t bytes) -> char* {
    char* p = ws + off;
    off += (bytes + 255) & ~(size_t)255;
    return p;
  };
  __hip_bfloat16* X0   = (__hip_bfloat16*)carve((size_t)SS * BB * EPAD * 2);
  __hip_bfloat16* WIH0 = (__hip_bfloat16*)carve((size_t)2 * G4H * EPAD * 2);
  __hip_bfloat16* WHH0 = (__hip_bfloat16*)carve((size_t)2 * G4H * HH * 2);
  __hip_bfloat16* WIH1 = (__hip_bfloat16*)carve((size_t)2 * G4H * (2 * HH) * 2);
  __hip_bfloat16* WHH1 = (__hip_bfloat16*)carve((size_t)2 * G4H * HH * 2);
  float* BIAS0 = (float*)carve((size_t)2 * G4H * 4);
  float* BIAS1 = (float*)carve((size_t)2 * G4H * 4);
  __hip_bfloat16* H0 = (__hip_bfloat16*)carve((size_t)SS * BB * (2 * HH) * 2);
  __hip_bfloat16* H1 = (__hip_bfloat16*)carve((size_t)SS * BB * (2 * HH) * 2);
  float* CST  = (float*)carve((size_t)2 * BB * HH * 4);
  float* EMIT = (float*)carve((size_t)BB * SS * TT * 4);
  int*   BAR  = (int*)carve(64);

  const int TPB = 256;

  // barrier init (needed each call: harness poisons ws once, replays graph)
  init_bar_kernel<<<1, 32, 0, stream>>>(BAR);

  // pack inputs to bf16 WMMA-friendly layouts
  {
    int tot = SS * BB * EPAD;
    pack_x0_kernel<<<(tot + TPB - 1) / TPB, TPB, 0, stream>>>(word_emb, X0);
  }
  {
    int tot = 2 * G4H * EPAD;
    pack_rows_kernel<<<(tot + TPB - 1) / TPB, TPB, 0, stream>>>(
        wih0, WIH0, 2 * G4H, EEIN, EPAD);
  }
  {
    int tot = 2 * G4H * HH;
    pack_rows_kernel<<<(tot + TPB - 1) / TPB, TPB, 0, stream>>>(
        whh0, WHH0, 2 * G4H, HH, HH);
  }
  {
    int tot = 2 * G4H * (2 * HH);
    pack_rows_kernel<<<(tot + TPB - 1) / TPB, TPB, 0, stream>>>(
        wih1, WIH1, 2 * G4H, 2 * HH, 2 * HH);
  }
  {
    int tot = 2 * G4H * HH;
    pack_rows_kernel<<<(tot + TPB - 1) / TPB, TPB, 0, stream>>>(
        whh1, WHH1, 2 * G4H, HH, HH);
  }
  bias_sum_kernel<<<(2 * G4H + TPB - 1) / TPB, TPB, 0, stream>>>(
      bih0, bhh0, BIAS0, 2 * G4H);
  bias_sum_kernel<<<(2 * G4H + TPB - 1) / TPB, TPB, 0, stream>>>(
      bih1, bhh1, BIAS1, 2 * G4H);

  // BiLSTM layer 0: X0 (K=320) -> H0
  lstm_layer_kernel<<<16, 256, 0, stream>>>(X0, EPAD, WIH0, WHH0, BIAS0, H0,
                                            CST, BAR);
  // BiLSTM layer 1: H0 (K=1024) -> H1
  lstm_layer_kernel<<<16, 256, 0, stream>>>(H0, 2 * HH, WIH1, WHH1, BIAS1, H1,
                                            CST, BAR);

  // projection to emissions
  {
    int tot = BB * SS * TT;
    proj_kernel<<<(tot + TPB - 1) / TPB, TPB, 0, stream>>>(H1, pw, pb, EMIT);
  }

  // CRF negative log-likelihood -> scalar
  crf_kernel<<<1, 32, 0, stream>>>(EMIT, tags, amask, trans, startt, endt, out);
}
